// Head_36438502540007
// MI455X (gfx1250) — matrix operand order repaired and checked by reading, not compiled, not run on hardware
//
#include <hip/hip_runtime.h>

typedef __attribute__((ext_vector_type(16))) __bf16 v16bf;
typedef __attribute__((ext_vector_type(8)))  __bf16 v8bf;
typedef __attribute__((ext_vector_type(2)))  __bf16 v2bf;
typedef __attribute__((ext_vector_type(8)))  float  v8f;

#define T_DIM 256
#define C_DIM 384
#define H_DIM 64
#define QK_PITCH 72      /* bf16/row for q,k: 64+8 pad */
#define VT_PITCH 264     /* bf16/row for vT and P strips: 256+8 pad */
#define P_PITCH  264
#define W_PITCH  392     /* bf16/row for staged W: 384+8 pad */

#define OFF_Q   0
#define OFF_K   (T_DIM * QK_PITCH * 2)             /* 36864  */
#define OFF_VT  (OFF_K + T_DIM * QK_PITCH * 2)     /* 73728  */
#define OFF_WP  (OFF_VT + H_DIM * VT_PITCH * 2)    /* 107520: union { W bf16 (phase1) | P strips (phase2) } */
#define W_BYTES (3 * H_DIM * W_PITCH * 2)          /* 150528 */
#define SMEM_BYTES (OFF_WP + W_BYTES)              /* 258048 */

/* q is pre-scaled by 1/sqrt(64) * log2(e) so softmax uses exp2 directly (exact base change) */
#define QSCALE (0.125f * 1.44269504088896340736f)

static __device__ __forceinline__ v8f wmma_bf16(v16bf a, v16bf b, v8f c) {
  return __builtin_amdgcn_wmma_f32_16x16x32_bf16(false, a, false, b, (short)0, c, false, false);
}

// A-operand from LDS: 16 contiguous bf16 at p. p 16B-aligned.
static __device__ __forceinline__ v16bf lds_ldA(const __bf16* p) {
  v8bf lo = *(const v8bf*)p;
  v8bf hi = *(const v8bf*)(p + 8);
  return __builtin_shufflevector(lo, hi, 0,1,2,3,4,5,6,7,8,9,10,11,12,13,14,15);
}
// B-operand from LDS: 16 contiguous bf16 at p. p 16B-aligned.
static __device__ __forceinline__ v16bf lds_ldB(const __bf16* p) {
  v8bf lo = *(const v8bf*)p;
  v8bf hi = *(const v8bf*)(p + 8);
  return __builtin_shufflevector(lo, hi, 0,1,2,3,4,5,6,7,8,9,10,11,12,13,14,15);
}
// A-operand from global fp32: elems 0..15 = p[0..15]
static __device__ __forceinline__ v16bf ldg_A_f32(const float* p) {
  float4 a0 = *(const float4*)(p);
  float4 a1 = *(const float4*)(p + 4);
  float4 a2 = *(const float4*)(p + 8);
  float4 a3 = *(const float4*)(p + 12);
  v16bf r;
  r[0]=(__bf16)a0.x;  r[1]=(__bf16)a0.y;  r[2]=(__bf16)a0.z;  r[3]=(__bf16)a0.w;
  r[4]=(__bf16)a1.x;  r[5]=(__bf16)a1.y;  r[6]=(__bf16)a1.z;  r[7]=(__bf16)a1.w;
  r[8]=(__bf16)a2.x;  r[9]=(__bf16)a2.y;  r[10]=(__bf16)a2.z; r[11]=(__bf16)a2.w;
  r[12]=(__bf16)a3.x; r[13]=(__bf16)a3.y; r[14]=(__bf16)a3.z; r[15]=(__bf16)a3.w;
  return r;
}

__global__ void __launch_bounds__(256)
attn_head_kernel(const float* __restrict__ logits, const float* __restrict__ Wq,
                 const float* __restrict__ Wk, const float* __restrict__ Wv,
                 float* __restrict__ out) {
  extern __shared__ char smem[];
  __bf16* qs   = (__bf16*)(smem + OFF_Q);
  __bf16* ks   = (__bf16*)(smem + OFF_K);
  __bf16* vts  = (__bf16*)(smem + OFF_VT);
  __bf16* wlds = (__bf16*)(smem + OFF_WP);   // phase-1 only; aliased by P strips in phase 2

  const int b    = blockIdx.x;
  const int tid  = threadIdx.x;
  const int lane = tid & 31;
  const int wave = tid >> 5;
  const int lh   = lane & 15;
  const int hi   = lane >> 4;
  const int k0a  = hi ? 16 : 0;
  const int k0b  = hi ? 16 : 0;

  // ---------------- Phase 0: stage W (Wq|Wk|Wv) into LDS as bf16 ----------------
  for (int c = tid; c < 192 * 96; c += 256) {
    const int row = c / 96;
    const int c4  = (c % 96) * 4;
    const float* src = (row < 64) ? Wq : (row < 128 ? Wk : Wv);
    const float4 x = *(const float4*)(src + (size_t)(row & 63) * C_DIM + c4);
    __bf16* dst = wlds + (size_t)row * W_PITCH + c4;
    dst[0] = (__bf16)x.x; dst[1] = (__bf16)x.y; dst[2] = (__bf16)x.z; dst[3] = (__bf16)x.w;
  }
  __syncthreads();

  // ---------------- Phase 1: q,k,v projections into LDS (bf16) ----------------
  const __bf16* wrow = wlds + lh * W_PITCH + k0b;   // tile t: row offset (t>>2)*64 + (t&3)*16
#pragma unroll
  for (int sel = 0; sel < 2; ++sel) {
    const int rt = wave + sel * 8;
    v8f acc[12];
#pragma unroll
    for (int t = 0; t < 12; ++t) acc[t] = (v8f){};

    const float* Abase = logits + (size_t)b * T_DIM * C_DIM + (size_t)(16 * rt + lh) * C_DIM;
    for (int kc = 0; kc < C_DIM / 32; ++kc) {
      v16bf a = ldg_A_f32(Abase + 32 * kc + k0a);
      // depth-2 software pipeline on B tiles
      v16bf bA = lds_ldB(wrow + 32 * kc);
      v16bf bB = lds_ldB(wrow + (size_t)16 * W_PITCH + 32 * kc);
#pragma unroll
      for (int t = 0; t < 12; ++t) {
        v16bf bC = bB;
        if (t < 10) bC = lds_ldB(wrow + (size_t)((t + 2) >> 2) * 64 * W_PITCH
                                      + (size_t)(((t + 2) & 3) * 16) * W_PITCH + 32 * kc);
        acc[t] = wmma_bf16(a, bA, acc[t]);
        bA = bB; bB = bC;
      }
    }
    // write q (pre-scaled), k row-major; v transposed with packed pair stores
#pragma unroll
    for (int jh = 0; jh < 4; ++jh) {
      const int h = 16 * jh + lh;
#pragma unroll
      for (int r = 0; r < 8; ++r) {
        const int t = 16 * rt + r + 8 * hi;
        qs[t * QK_PITCH + h] = (__bf16)(acc[jh][r] * QSCALE);
        ks[t * QK_PITCH + h] = (__bf16)acc[4 + jh][r];
      }
#pragma unroll
      for (int r = 0; r < 8; r += 2) {
        const int t = 16 * rt + r + 8 * hi;
        v2bf p; p[0] = (__bf16)acc[8 + jh][r]; p[1] = (__bf16)acc[8 + jh][r + 1];
        *(v2bf*)(vts + h * VT_PITCH + t) = p;
      }
    }
  }
  __syncthreads();   // also retires all W reads before P strips overwrite the region

  // ---------------- Phase 2: causal softmax attention ----------------
  __bf16* Pw = (__bf16*)(smem + OFF_WP) + wave * 16 * P_PITCH;

#pragma unroll
  for (int sel = 0; sel < 2; ++sel) {
    const int it = sel ? (15 - wave) : wave;

    // q A-operands are invariant over j: hoist
    const __bf16* qbase = qs + (16 * it + lh) * QK_PITCH + k0a;
    const v16bf aq0 = lds_ldA(qbase);
    const v16bf aq1 = lds_ldA(qbase + 32);
    const __bf16* kbase = ks + lh * QK_PITCH + k0b;

    // ---- pass 1: per-lane running max over tiles; one shuffle reduction at end ----
    float rmp[8];
#pragma unroll
    for (int r = 0; r < 8; ++r) rmp[r] = -3.0e38f;
    {
      const int j1 = (it > 0) ? 1 : 0;
      v16bf kb0a = lds_ldB(kbase);
      v16bf kb1a = lds_ldB(kbase + 32);
      v16bf kb0b = lds_ldB(kbase + (size_t)(16 * j1) * QK_PITCH);
      v16bf kb1b = lds_ldB(kbase + (size_t)(16 * j1) * QK_PITCH + 32);
      for (int j = 0; j <= it; ++j) {
        const int j2 = (j + 2 <= it) ? j + 2 : it;
        v8f s = {};
        s = wmma_bf16(aq0, kb0a, s);
        v16bf n0 = lds_ldB(kbase + (size_t)(16 * j2) * QK_PITCH);
        v16bf n1 = lds_ldB(kbase + (size_t)(16 * j2) * QK_PITCH + 32);
        s = wmma_bf16(aq1, kb1a, s);
        if (j == it) {
#pragma unroll
          for (int r = 0; r < 8; ++r)
            if (lh > r + 8 * hi) s[r] = -1e30f;
        }
#pragma unroll
        for (int r = 0; r < 8; ++r) rmp[r] = fmaxf(rmp[r], s[r]);
        kb0a = kb0b; kb1a = kb1b; kb0b = n0; kb1b = n1;
      }
    }
    float rm[8];
#pragma unroll
    for (int r = 0; r < 8; ++r) {
      float m = rmp[r];
      for (int d = 1; d < 16; d <<= 1) m = fmaxf(m, __shfl_xor(m, d, 16));
      rm[r] = m;
    }

    // ---- pass 2: exp2, write P (bf16), per-lane partial sums; reduce once at end ----
    float rsp[8];
#pragma unroll
    for (int r = 0; r < 8; ++r) rsp[r] = 0.0f;
    {
      const int j1 = (it > 0) ? 1 : 0;
      v16bf kb0a = lds_ldB(kbase);
      v16bf kb1a = lds_ldB(kbase + 32);
      v16bf kb0b = lds_ldB(kbase + (size_t)(16 * j1) * QK_PITCH);
      v16bf kb1b = lds_ldB(kbase + (size_t)(16 * j1) * QK_PITCH + 32);
      for (int j = 0; j <= it; ++j) {
        const int j2 = (j + 2 <= it) ? j + 2 : it;
        v8f s = {};
        s = wmma_bf16(aq0, kb0a, s);
        v16bf n0 = lds_ldB(kbase + (size_t)(16 * j2) * QK_PITCH);
        v16bf n1 = lds_ldB(kbase + (size_t)(16 * j2) * QK_PITCH + 32);
        s = wmma_bf16(aq1, kb1a, s);
        if (j == it) {
#pragma unroll
          for (int r = 0; r < 8; ++r)
            if (lh > r + 8 * hi) s[r] = -1e30f;
        }
#pragma unroll
        for (int r = 0; r < 8; ++r) {
          float e = __builtin_amdgcn_exp2f(s[r] - rm[r]);
          Pw[(r + 8 * hi) * P_PITCH + 16 * j + lh] = (__bf16)e;
          rsp[r] += e;
        }
        kb0a = kb0b; kb1a = kb1b; kb0b = n0; kb1b = n1;
      }
    }
    float inv[8];
#pragma unroll
    for (int r = 0; r < 8; ++r) {
      float t = rsp[r];
      for (int d = 1; d < 16; d <<= 1) t += __shfl_xor(t, d, 16);
      inv[r] = 1.0f / t;
    }

    // pad P cols to a multiple of 32
    int ntile = it + 1;
    if (ntile & 1) {
      v8bf z;
#pragma unroll
      for (int e = 0; e < 8; ++e) z[e] = (__bf16)0.0f;
      *(v8bf*)(Pw + lh * P_PITCH + ntile * 16 + hi * 8) = z;
      ++ntile;
    }
    asm volatile("s_wait_dscnt 0" ::: "memory");   // same-wave cross-lane LDS write->read

    // ---- pass 3: out[16,64] = P * V (full next-iteration prefetch) ----
    v8f o[4];
#pragma unroll
    for (int jh = 0; jh < 4; ++jh) o[jh] = (v8f){};
    const int nkk = ntile >> 1;
    {
      const __bf16* pp = Pw + lh * P_PITCH + k0a;
      const __bf16* vp = vts + lh * VT_PITCH + k0b;
      v16bf pa = lds_ldA(pp);
      v16bf c0 = lds_ldB(vp);
      v16bf c1 = lds_ldB(vp + (size_t)16 * VT_PITCH);
      v16bf c2 = lds_ldB(vp + (size_t)32 * VT_PITCH);
      v16bf c3 = lds_ldB(vp + (size_t)48 * VT_PITCH);
      for (int kk = 0; kk < nkk; ++kk) {
        const int kn = (kk + 1 < nkk) ? kk + 1 : kk;
        v16bf pan = lds_ldA(pp + 32 * kn);
        v16bf n0 = lds_ldB(vp + (size_t)0  * VT_PITCH + 32 * kn);
        v16bf n1 = lds_ldB(vp + (size_t)16 * VT_PITCH + 32 * kn);
        v16bf n2 = lds_ldB(vp + (size_t)32 * VT_PITCH + 32 * kn);
        v16bf n3 = lds_ldB(vp + (size_t)48 * VT_PITCH + 32 * kn);
        o[0] = wmma_bf16(pa, c0, o[0]);
        o[1] = wmma_bf16(pa, c1, o[1]);
        o[2] = wmma_bf16(pa, c2, o[2]);
        o[3] = wmma_bf16(pa, c3, o[3]);
        pa = pan; c0 = n0; c1 = n1; c2 = n2; c3 = n3;
      }
    }

    float* outp = out + ((size_t)b * T_DIM + 16 * it) * H_DIM;
#pragma unroll
    for (int jh = 0; jh < 4; ++jh) {
#pragma unroll
      for (int r = 0; r < 8; ++r) {
        outp[(r + 8 * hi) * H_DIM + 16 * jh + lh] = o[jh][r] * inv[r];
      }
    }
  }
}

extern "C" void kernel_launch(void* const* d_in, const int* in_sizes, int n_in,
                              void* d_out, int out_size, void* d_ws, size_t ws_size,
                              hipStream_t stream) {
  (void)n_in; (void)out_size; (void)d_ws; (void)ws_size;
  const float* logits = (const float*)d_in[0];
  const float* Wq     = (const float*)d_in[1];
  const float* Wk     = (const float*)d_in[2];
  const float* Wv     = (const float*)d_in[3];
  float* out = (float*)d_out;

  const int B = in_sizes[0] / (T_DIM * C_DIM);   // 512

  hipFuncSetAttribute((const void*)attn_head_kernel,
                      hipFuncAttributeMaxDynamicSharedMemorySize, SMEM_BYTES);
  attn_head_kernel<<<dim3(B), dim3(256), SMEM_BYTES, stream>>>(logits, Wq, Wk, Wv, out);
}
